// Gemma2Attention_5016521801817
// MI455X (gfx1250) — matrix operand order repaired and checked by reading, not compiled, not run on hardware
//
#include <hip/hip_runtime.h>
#include <stdint.h>

#define B_ 4
#define S_ 2048
#define H_ 2048
#define NH_ 8
#define NKV_ 4
#define HD_ 256
#define WINDOW_ 1024
#define SCALE_ 0.0625f   // 256^-0.5

typedef __attribute__((ext_vector_type(16))) __bf16 v16bf;
typedef __attribute__((ext_vector_type(8)))  float  v8f;

union FragAB {
  v16bf v;
  unsigned short s[16];
  unsigned int   u[8];
  uint4          q4[2];
};

__device__ __forceinline__ unsigned short f2bf(float x) {
  unsigned int b = __float_as_uint(x);
  unsigned int r = b + 0x7FFFu + ((b >> 16) & 1u);  // RNE
  return (unsigned short)(r >> 16);
}
__device__ __forceinline__ float bf2f(unsigned short h) {
  return __uint_as_float(((unsigned int)h) << 16);
}
__device__ __forceinline__ float rmax16(float x) {
  #pragma unroll
  for (int off = 1; off < 16; off <<= 1) x = fmaxf(x, __shfl_xor(x, off, 32));
  return x;
}
__device__ __forceinline__ float rsum16(float x) {
  #pragma unroll
  for (int off = 1; off < 16; off <<= 1) x += __shfl_xor(x, off, 32);
  return x;
}

// ---------------- Tensor Data Mover (guarded) ----------------
#if defined(__gfx1250__) && __has_builtin(__builtin_amdgcn_tensor_load_to_lds)
#define HAVE_TDM 1
typedef __attribute__((ext_vector_type(4))) unsigned int u32x4;
typedef __attribute__((ext_vector_type(8))) int          i32x8;
typedef __attribute__((ext_vector_type(4))) int          i32x4;

// 2-D tile load: tile_d0 = contiguous elements per row, tile_d1 = rows,
// stride = elements between rows (data_size = 2 bytes).
__device__ __forceinline__ void tdm_load_2d(unsigned lds_addr, const void* gptr,
                                            unsigned tile_d0, unsigned tile_d1,
                                            unsigned long long stride_elems) {
  unsigned long long ga = (unsigned long long)(uintptr_t)gptr;
  u32x4 g0;
  g0[0] = 1u;                                            // count=1 (user D#)
  g0[1] = lds_addr;                                      // lds_addr [63:32]
  g0[2] = (unsigned)(ga & 0xFFFFFFFFu);                  // global_addr[31:0]
  g0[3] = (unsigned)((ga >> 32) & 0x01FFFFFFu) | (2u << 30);  // addr[56:32], type=2
  i32x8 g1;
  g1[0] = (int)(1u << 16);          // workgroup_mask=0, data_size=1 (2 bytes)
  g1[1] = (int)(tile_d0 << 16);     // tensor_dim0[15:0] @ bits 63:48
  g1[2] = (int)(tile_d1 << 16);     // tensor_dim0 hi=0, tensor_dim1[15:0] @ 95:80
  g1[3] = (int)(tile_d0 << 16);     // tensor_dim1 hi=0, tile_dim0 @ 127:112
  g1[4] = (int)tile_d1;             // tile_dim1 @ 143:128, tile_dim2=0
  g1[5] = (int)(unsigned)(stride_elems & 0xFFFFFFFFu);   // dim0_stride lo
  g1[6] = (int)(unsigned)((stride_elems >> 32) & 0xFFFFu); // dim0_stride hi
  g1[7] = 0;
  i32x4 z4 = {0, 0, 0, 0};
#if __clang_major__ >= 23
  i32x8 z8 = {0, 0, 0, 0, 0, 0, 0, 0};
  __builtin_amdgcn_tensor_load_to_lds(g0, g1, z4, z4, z8, 0);
#else
  __builtin_amdgcn_tensor_load_to_lds(g0, g1, z4, z4, 0);
#endif
}
#else
#define HAVE_TDM 0
#endif

// ---------------- f32 -> bf16 conversion ----------------
__global__ void cvt_f32_bf16(const float* __restrict__ in,
                             unsigned short* __restrict__ out, int n) {
  int i = blockIdx.x * blockDim.x + threadIdx.x;
  if (i < n) out[i] = f2bf(in[i]);
}

// ---------------- f32 [K][N] -> bf16 transposed [N][K] ----------------
__global__ __launch_bounds__(256)
void cvt_transpose(const float* __restrict__ in, unsigned short* __restrict__ out,
                   int K, int N) {
  __shared__ float tile[32][33];
  int k0 = blockIdx.y * 32, n0 = blockIdx.x * 32;
  int tx = threadIdx.x & 31, ty = threadIdx.x >> 5;   // 32 x 8
  #pragma unroll
  for (int i = ty; i < 32; i += 8)
    tile[i][tx] = in[(size_t)(k0 + i) * N + n0 + tx];
  __syncthreads();
  #pragma unroll
  for (int i = ty; i < 32; i += 8)
    out[(size_t)(n0 + i) * K + k0 + tx] = f2bf(tile[tx][i]);
}

// ---------------- bf16 V [B,S,NKV,HD] -> V^T [B,NKV,HD,S] ----------------
__global__ __launch_bounds__(256)
void transpose_v(const unsigned short* __restrict__ in,
                 unsigned short* __restrict__ out) {
  __shared__ unsigned short tile[32][33];
  int bkv = blockIdx.z;
  int b = bkv / NKV_, kv = bkv % NKV_;
  int s0 = blockIdx.y * 32, d0 = blockIdx.x * 32;
  int tx = threadIdx.x & 31, ty = threadIdx.x >> 5;
  #pragma unroll
  for (int i = ty; i < 32; i += 8)
    tile[i][tx] = in[((size_t)(b * S_ + s0 + i) * NKV_ + kv) * HD_ + d0 + tx];
  __syncthreads();
  #pragma unroll
  for (int i = ty; i < 32; i += 8)
    out[((size_t)(b * NKV_ + kv) * HD_ + d0 + i) * S_ + s0 + tx] = tile[tx][i];
}

// ---------------- bf16 GEMM: C[M,N] = A[M,K] * Bt[N,K]^T (+bias) ----------------
// block: 256 threads = 8 waves, tile 128x128, K-step 32.
__global__ __launch_bounds__(256)
void gemm_bf16(const unsigned short* __restrict__ A,
               const unsigned short* __restrict__ Bt,   // [N][K]
               const float* __restrict__ bias,
               void* __restrict__ Cout,
               int M, int N, int K, int outBf16) {
  __shared__ __align__(16) unsigned short sA[128 * 32];
  __shared__ __align__(16) unsigned short sB[128 * 32];  // [n][k]
  const int tid  = threadIdx.x;
  const int lane = tid & 31;
  const int wave = tid >> 5;
  const int waveM = wave >> 2;      // 0..1
  const int waveN = wave & 3;       // 0..3
  const int rowBase = blockIdx.y * 128;
  const int colBase = blockIdx.x * 128;
  const int lm = lane & 15;
  const int lg = lane >> 4;

  v8f zero = {};
  v8f acc[4][2];
  #pragma unroll
  for (int f = 0; f < 4; ++f)
    #pragma unroll
    for (int n = 0; n < 2; ++n) acc[f][n] = zero;

  for (int k0 = 0; k0 < K; k0 += 32) {
    __syncthreads();
    #pragma unroll
    for (int c = tid; c < 512; c += 256) {
      int r  = c >> 2;
      int co = (c & 3) << 3;
      *(uint4*)&sA[r * 32 + co] =
          *(const uint4*)&A[(size_t)(rowBase + r) * K + k0 + co];
      *(uint4*)&sB[r * 32 + co] =
          *(const uint4*)&Bt[(size_t)(colBase + r) * K + k0 + co];
    }
    __syncthreads();

    FragAB a[4], b[2];
    #pragma unroll
    for (int f = 0; f < 4; ++f) {
      int m = waveM * 64 + f * 16 + lm;
      a[f].q4[0] = *(const uint4*)&sA[m * 32 + 8 * lg];
      a[f].q4[1] = *(const uint4*)&sA[m * 32 + 16 + 8 * lg];
    }
    #pragma unroll
    for (int n = 0; n < 2; ++n) {
      int col = waveN * 32 + n * 16 + lm;
      const unsigned short* src = &sB[col * 32 + 8 * lg];
      b[n].q4[0] = ((const uint4*)src)[0];
      b[n].q4[1] = ((const uint4*)src)[2];
    }
    #pragma unroll
    for (int f = 0; f < 4; ++f)
      #pragma unroll
      for (int n = 0; n < 2; ++n)
        acc[f][n] = __builtin_amdgcn_wmma_f32_16x16x32_bf16(
            false, a[f].v, false, b[n].v, (short)0, acc[f][n], false, false);
  }

  #pragma unroll
  for (int f = 0; f < 4; ++f)
    #pragma unroll
    for (int n = 0; n < 2; ++n)
      #pragma unroll
      for (int r = 0; r < 8; ++r) {
        int row = rowBase + waveM * 64 + f * 16 + r + 8 * lg;
        int col = colBase + waveN * 32 + n * 16 + lm;
        float v = acc[f][n][r];
        if (bias) v += bias[col];
        if (outBf16)
          ((unsigned short*)Cout)[(size_t)row * N + col] = f2bf(v);
        else
          ((float*)Cout)[(size_t)row * N + col] = v;
      }
}

// ---------------- NeoX RoPE (in-place on bf16 [B,S,nheads,HD]) ----------------
__global__ void rope_kernel(unsigned short* __restrict__ x,
                            const int* __restrict__ pos, int nheads) {
  const int half = HD_ / 2;
  int idx = blockIdx.x * blockDim.x + threadIdx.x;
  int total = B_ * S_ * nheads * half;
  if (idx >= total) return;
  int i = idx % half;
  int h = (idx / half) % nheads;
  int s = (idx / (half * nheads)) % S_;
  int b = idx / (half * nheads * S_);
  size_t base = ((size_t)(b * S_ + s) * nheads + h) * HD_;
  float x1 = bf2f(x[base + i]);
  float x2 = bf2f(x[base + i + half]);
  float p  = (float)pos[b * S_ + s];
  float inv = __expf(-((float)i / (float)half) * 9.210340371976184f); // ln(1e4)
  float f = p * inv;
  float sn, cs;
  __sincosf(f, &sn, &cs);
  x[base + i]        = f2bf(x1 * cs - x2 * sn);
  x[base + i + half] = f2bf(x2 * cs + x1 * sn);
}

// ---------------- flash attention (sliding-window causal, GQA) ----------------
// grid: (S/64, B*NH), block: 128 (4 waves). Each wave owns a 16-row q tile.
// K chunk staged row-major [key][d]; V chunk staged transposed [d][key].
__global__ __launch_bounds__(128)
void attn_kernel(const unsigned short* __restrict__ q,
                 const unsigned short* __restrict__ k,
                 const unsigned short* __restrict__ vt,  // [B,NKV,HD,S]
                 unsigned short* __restrict__ o) {
  __shared__ __align__(16) unsigned short sK[32 * HD_];   // [key][d]
  __shared__ __align__(16) unsigned short sVt[HD_ * 32];  // [d][key]
  __shared__ __align__(16) unsigned short sP[4][16 * 32];

  const int tid  = threadIdx.x;
  const int lane = tid & 31;
  const int wave = tid >> 5;
  const int lm = lane & 15;
  const int lg = lane >> 4;
  const int bh = blockIdx.y;
  const int b  = bh / NH_;
  const int h  = bh % NH_;
  const int kvh = h / (NH_ / NKV_);
  const int q0b = blockIdx.x * 64;
  const int q0  = q0b + wave * 16;

  // Q fragments: 8 A-frags (16x32 over HD=256), in registers.
  FragAB qa[8];
  {
    size_t qrow = ((size_t)(b * S_ + q0 + lm) * NH_ + h) * HD_;
    #pragma unroll
    for (int dk = 0; dk < 8; ++dk) {
      qa[dk].q4[0] = *(const uint4*)&q[qrow + dk * 32 + 8 * lg];
      qa[dk].q4[1] = *(const uint4*)&q[qrow + dk * 32 + 16 + 8 * lg];
    }
  }

  v8f zero = {};
  v8f acc[16];
  #pragma unroll
  for (int t = 0; t < 16; ++t) acc[t] = zero;
  float mrun[8], lrun[8];
  #pragma unroll
  for (int r = 0; r < 8; ++r) { mrun[r] = -1e30f; lrun[r] = 0.0f; }

  int jlo = q0b - (WINDOW_ - 1);
  if (jlo < 0) jlo = 0;
  const int c_lo = (jlo / 32) * 32;
  const int c_hi = q0b + 63;

#if HAVE_TDM
  const unsigned ldsK = (unsigned)(uintptr_t)(void*)&sK[0];
  const unsigned ldsV = (unsigned)(uintptr_t)(void*)&sVt[0];
#endif

  for (int jc = c_lo; jc <= c_hi; jc += 32) {
    __syncthreads();   // previous chunk fully consumed
#if HAVE_TDM
    if (wave == 0) {
      // K tile: 32 rows of 256 contiguous bf16, row stride NKV*HD elements.
      tdm_load_2d(ldsK, k + ((size_t)(b * S_ + jc) * NKV_ + kvh) * HD_,
                  HD_, 32, (unsigned long long)(NKV_ * HD_));
      // V^T tile: 256 rows of 32 contiguous bf16, row stride S elements.
      tdm_load_2d(ldsV, vt + ((size_t)(b * NKV_ + kvh) * HD_) * S_ + jc,
                  32, HD_, (unsigned long long)S_);
      __builtin_amdgcn_s_wait_tensorcnt(0);
    }
#else
    #pragma unroll
    for (int ii = 0; ii < 8; ++ii) {
      int lin = tid + 128 * ii;        // 0..1023
      int rk = lin >> 5, ck = lin & 31;
      *(uint4*)&sK[rk * HD_ + ck * 8] =
          *(const uint4*)&k[((size_t)(b * S_ + jc + rk) * NKV_ + kvh) * HD_ + ck * 8];
      int rv = lin >> 2, cv = lin & 3;
      *(uint4*)&sVt[rv * 32 + cv * 8] =
          *(const uint4*)&vt[((size_t)(b * NKV_ + kvh) * HD_ + rv) * S_ + jc + cv * 8];
    }
#endif
    __syncthreads();

    // scores: two 16-key N-tiles
    float sc[2][8];
    #pragma unroll
    for (int nt = 0; nt < 2; ++nt) {
      v8f s = zero;
      #pragma unroll
      for (int dk = 0; dk < 8; ++dk) {
        FragAB kb;
        const unsigned short* src = &sK[(nt * 16 + lm) * HD_ + dk * 32 + 8 * lg];
        kb.q4[0] = ((const uint4*)src)[0];
        kb.q4[1] = ((const uint4*)src)[2];
        s = __builtin_amdgcn_wmma_f32_16x16x32_bf16(
            false, qa[dk].v, false, kb.v, (short)0, s, false, false);
      }
      #pragma unroll
      for (int r = 0; r < 8; ++r) {
        int irow = q0 + r + 8 * lg;
        int jcol = jc + nt * 16 + lm;
        bool ok = (jcol <= irow) && (jcol > irow - WINDOW_);
        sc[nt][r] = ok ? s[r] * SCALE_ : -1e30f;
      }
    }

    // online softmax update
    float corr[8];
    #pragma unroll
    for (int r = 0; r < 8; ++r) {
      float tm = rmax16(fmaxf(sc[0][r], sc[1][r]));
      float mnew = fmaxf(mrun[r], tm);
      corr[r] = __expf(mrun[r] - mnew);
      float p0 = (sc[0][r] > -1e29f) ? __expf(sc[0][r] - mnew) : 0.0f;
      float p1 = (sc[1][r] > -1e29f) ? __expf(sc[1][r] - mnew) : 0.0f;
      sP[wave][(r + 8 * lg) * 32 + lm]      = f2bf(p0);
      sP[wave][(r + 8 * lg) * 32 + 16 + lm] = f2bf(p1);
      lrun[r] = lrun[r] * corr[r] + rsum16(p0 + p1);
      mrun[r] = mnew;
    }
    #pragma unroll
    for (int t = 0; t < 16; ++t)
      #pragma unroll
      for (int r = 0; r < 8; ++r) acc[t][r] *= corr[r];
    __syncthreads();   // sP visible

    // P (A-layout) x V
    FragAB pa;
    pa.q4[0] = *(const uint4*)&sP[wave][lm * 32 + 8 * lg];
    pa.q4[1] = *(const uint4*)&sP[wave][lm * 32 + 16 + 8 * lg];
    #pragma unroll
    for (int t16 = 0; t16 < 16; ++t16) {
      FragAB vb;
      const unsigned short* src = &sVt[(t16 * 16 + lm) * 32 + 8 * lg];
      vb.q4[0] = ((const uint4*)src)[0];
      vb.q4[1] = ((const uint4*)src)[2];
      acc[t16] = __builtin_amdgcn_wmma_f32_16x16x32_bf16(
          false, pa.v, false, vb.v, (short)0, acc[t16], false, false);
    }
  }

  // normalize + store bf16 [B,S,NH,HD]
  #pragma unroll
  for (int t = 0; t < 16; ++t)
    #pragma unroll
    for (int r = 0; r < 8; ++r) {
      int m = r + 8 * lg;
      float val = acc[t][r] / lrun[r];
      size_t off = ((size_t)(b * S_ + q0 + m) * NH_ + h) * HD_ + t * 16 + lm;
      o[off] = f2bf(val);
    }
}

// ---------------- host launch ----------------
extern "C" void kernel_launch(void* const* d_in, const int* in_sizes, int n_in,
                              void* d_out, int out_size, void* d_ws, size_t ws_size,
                              hipStream_t stream) {
  (void)in_sizes; (void)n_in; (void)out_size; (void)ws_size;
  const float* hs = (const float*)d_in[0];
  const int*  pos = (const int*)d_in[1];
  const float* Wq = (const float*)d_in[2];
  const float* bq = (const float*)d_in[3];
  const float* Wk = (const float*)d_in[4];
  const float* bk = (const float*)d_in[5];
  const float* Wv = (const float*)d_in[6];
  const float* bv = (const float*)d_in[7];
  const float* Wo = (const float*)d_in[8];
  float* out = (float*)d_out;

  char* ws = (char*)d_ws;
  size_t off = 0;
  auto alloc = [&](size_t bytes) -> void* {
    void* p = ws + off;
    off += (bytes + 255) & ~(size_t)255;
    return p;
  };
  const size_t M = (size_t)B_ * S_;          // 8192
  unsigned short* hsb  = (unsigned short*)alloc(M * H_ * 2);
  unsigned short* wqbt = (unsigned short*)alloc((size_t)H_ * (NH_ * HD_) * 2);
  unsigned short* wkbt = (unsigned short*)alloc((size_t)H_ * (NKV_ * HD_) * 2);
  unsigned short* wvbt = (unsigned short*)alloc((size_t)H_ * (NKV_ * HD_) * 2);
  unsigned short* wobt = (unsigned short*)alloc((size_t)(NH_ * HD_) * H_ * 2);
  unsigned short* qb   = (unsigned short*)alloc(M * NH_ * HD_ * 2);
  unsigned short* kb   = (unsigned short*)alloc(M * NKV_ * HD_ * 2);
  unsigned short* vb   = (unsigned short*)alloc(M * NKV_ * HD_ * 2);
  unsigned short* vtb  = (unsigned short*)alloc(M * NKV_ * HD_ * 2);
  unsigned short* ab   = (unsigned short*)alloc(M * NH_ * HD_ * 2);

  // hidden -> bf16 (row-major), weights -> bf16 transposed [N][K]
  cvt_f32_bf16<<<(int)((M * H_ + 255) / 256), 256, 0, stream>>>(hs, hsb, (int)(M * H_));
  cvt_transpose<<<dim3((NH_ * HD_) / 32, H_ / 32), 256, 0, stream>>>(Wq, wqbt, H_, NH_ * HD_);
  cvt_transpose<<<dim3((NKV_ * HD_) / 32, H_ / 32), 256, 0, stream>>>(Wk, wkbt, H_, NKV_ * HD_);
  cvt_transpose<<<dim3((NKV_ * HD_) / 32, H_ / 32), 256, 0, stream>>>(Wv, wvbt, H_, NKV_ * HD_);
  cvt_transpose<<<dim3(H_ / 32, (NH_ * HD_) / 32), 256, 0, stream>>>(Wo, wobt, NH_ * HD_, H_);

  // QKV projections (bf16 out)
  gemm_bf16<<<dim3((NH_ * HD_) / 128, M / 128), 256, 0, stream>>>(
      hsb, wqbt, bq, qb, (int)M, NH_ * HD_, H_, 1);
  gemm_bf16<<<dim3((NKV_ * HD_) / 128, M / 128), 256, 0, stream>>>(
      hsb, wkbt, bk, kb, (int)M, NKV_ * HD_, H_, 1);
  gemm_bf16<<<dim3((NKV_ * HD_) / 128, M / 128), 256, 0, stream>>>(
      hsb, wvbt, bv, vb, (int)M, NKV_ * HD_, H_, 1);

  // RoPE on q and k
  {
    int tq = B_ * S_ * NH_ * (HD_ / 2);
    rope_kernel<<<(tq + 255) / 256, 256, 0, stream>>>(qb, pos, NH_);
    int tk = B_ * S_ * NKV_ * (HD_ / 2);
    rope_kernel<<<(tk + 255) / 256, 256, 0, stream>>>(kb, pos, NKV_);
  }

  // V -> V^T [B,NKV,HD,S]
  transpose_v<<<dim3(HD_ / 32, S_ / 32, B_ * NKV_), 256, 0, stream>>>(vb, vtb);

  // flash attention
  attn_kernel<<<dim3(S_ / 64, B_ * NH_), 128, 0, stream>>>(qb, kb, vtb, ab);

  // output projection (f32 out, no bias)
  gemm_bf16<<<dim3(H_ / 128, M / 128), 256, 0, stream>>>(
      ab, wobt, nullptr, out, (int)M, H_, NH_ * HD_, 0);
}
